// CSWinMLPLayer_13735305412886
// MI455X (gfx1250) — matrix operand order repaired and checked by reading, not compiled, not run on hardware
//
#include <hip/hip_runtime.h>
#include <hip/hip_bf16.h>
#include <math.h>

// ---------------------------------------------------------------------------
// CSWin-style mixing-attention + MLP layer for MI455X (gfx1250, wave32).
// All large GEMMs run on v_wmma_f32_16x16x32_f16 (f16 in, f32 accumulate).
// ~265 GFLOP total, dominated by the 2048x416x38416 generate GEMMs and the
// 50176x384x1536 MLP GEMMs. g_w (30MB f16) is L2-resident (192MB L2).
// LDS tiles are stored FRAGMENT-MAJOR: each lane's 16 operand halves are two
// contiguous 16B runs -> 2x ds_load_b128 per fragment (no ds_load_u16 chains).
// Workspace ~576 MB (logits buffer 314MB reused for branch1 + MLP hidden).
// ---------------------------------------------------------------------------

typedef _Float16 half_t;
typedef __attribute__((ext_vector_type(16))) _Float16 v16h;
typedef __attribute__((ext_vector_type(8)))  _Float16 v8h;
typedef __attribute__((ext_vector_type(8)))  float    v8f;

#define BQ    64
#define RESO  28
#define NTOK  784            // 28*28
#define DIMC  384
#define CBCH  192            // per-branch channels
#define HEADS 8
#define GWIN  32             // windows*heads per batch per branch
#define WIN   196            // 28*7
#define WD    392            // win * DMIX
#define WDP   416            // WD padded to multiple of 32 for WMMA K
#define NGEN  38416          // 196*196
#define HID   1536
#define ROWS  (BQ*NTOK)      // 50176
#define GROWS (BQ*GWIN)      // 2048
#define JC    49             // softmax column chunk (196 = 4*49)

union AFrag { v16h v; half_t h[16]; v8h h8[2]; };
union CFrag { v8f  v; float  f[8];  };
union H8    { v8h  v; half_t h[8];  };

// ---------------------------------------------------------------------------
// Tiled WMMA GEMM: C(MxN,f32 acc) = A(MxK,f16) * B(KxN,f16) + bias
// mode 0: outF = acc + bias                         (logits)
// mode 1: outH = (f16) gelu_exact(acc + bias)       (fc1 + GELU)
// mode 2: outF = resid + alpha[col]*(acc + bias)    (proj / fc2 residual)
//
// Tile: BM=128, BN=128, BK=32. 256 threads = 8 waves in 4(M)x2(N); each wave
// owns 32x64 = 2(A-frag) x 4(B-frag) = 8 WMMAs per k-step.
//
// Fragment-major LDS:
//   A (16x32, pair-interleaved per ISA 7.12.2):
//     e = (k&7) + 8*(k>>4), lane = (m&15) + 16*((k>>3)&1)
//     -> aligned 8-wide k-chunk of a row is a contiguous 8-half lane run,
//        so global b128 -> single ds_store_b128; frag = 2x ds_load_b128.
//   B (32x16, same k order as A): lane = (n&15) + 16*((k>>3)&1), e = (k&7) + 8*(k>>4)
//     -> global b128 (8 cols of one k-row) scatters as 8 ds_store_b16;
//        frag = 2x ds_load_b128.
// M multiple of 128, K multiple of 32, N multiple of 8.
// ---------------------------------------------------------------------------
__global__ __launch_bounds__(256)
void gemm_wmma_f16(const half_t* __restrict__ A, int lda,
                   const half_t* __restrict__ Bm, int ldb,
                   const float* __restrict__ bias,
                   float* __restrict__ outF,
                   half_t* __restrict__ outH,
                   const float* __restrict__ resid,
                   const float* __restrict__ alpha,
                   int M, int N, int K, int mode)
{
  __shared__ half_t As[8][2][32][8];   // [mtile][e8][lane][j]  8 KB
  __shared__ half_t Bs[8][2][32][8];   // [ntile][e8][lane][j]  8 KB

  const int tid  = threadIdx.x;
  const int lane = tid & 31;
  const int wave = tid >> 5;
  const int wm   = wave >> 1;          // 0..3 : rows [wm*32, +32)
  const int wn   = wave & 1;           // 0..1 : cols [wn*64, +64)
  const int m0   = blockIdx.y * 128;
  const int n0   = blockIdx.x * 128;

  CFrag acc[2][4];
#pragma unroll
  for (int i = 0; i < 2; ++i)
#pragma unroll
    for (int j = 0; j < 4; ++j)
#pragma unroll
      for (int r = 0; r < 8; ++r) acc[i][j].f[r] = 0.0f;

  const int ktiles = K >> 5;
  for (int kt = 0; kt < ktiles; ++kt) {
    const int k0 = kt << 5;

    // ---- A tile: 128 rows x 32 halves = 512 b128 loads; 2 per thread.
    //      Store directly in fragment-major order (single b128 store each).
#pragma unroll
    for (int t = 0; t < 2; ++t) {
      const int id  = tid + t * 256;
      const int row = id >> 2;          // 0..127
      const int c   = id & 3;           // k chunk of 8
      const v8h src = *(const v8h*)(A + (size_t)(m0 + row) * lda + k0 + c * 8);
      *(v8h*)(&As[row >> 4][c >> 1][(row & 15) + 16 * (c & 1)][0]) = src;
    }
    // ---- B tile: 32 rows x 128 halves; 2 b128 loads per thread, each
    //      scattered to 8 per-lane fragment slots (transpose into frag-major).
#pragma unroll
    for (int t = 0; t < 2; ++t) {
      const int c    = tid + t * 256;
      const int row  = c >> 4;          // k row 0..31
      const int ch   = c & 15;          // 8-col chunk
      const int gc   = n0 + ch * 8;
      H8 val;
#pragma unroll
      for (int e = 0; e < 8; ++e) val.h[e] = (half_t)0.0f;
      if (gc < N) val.v = *(const v8h*)(Bm + (size_t)(k0 + row) * ldb + gc);
      const int ntile = ch >> 1;
      const int nb8   = (ch & 1) * 8;
      const int khi   = (row >> 3) & 1;
      const int e8    = row >> 4;
      const int jk    = row & 7;
#pragma unroll
      for (int jj = 0; jj < 8; ++jj)
        Bs[ntile][e8][nb8 + jj + 16 * khi][jk] = val.h[jj];
    }
    // prefetch next k-tile of B into GL2 (global_prefetch_b8)
    if (kt + 1 < ktiles) {
      const int gc = n0 + ((tid & 15) * 8);
      if (gc < N)
        __builtin_prefetch(Bm + (size_t)(k0 + 32 + (tid >> 4)) * ldb + gc, 0, 1);
    }
    __syncthreads();

    // ---- fragments: two contiguous 16B LDS loads each
    AFrag af[2];
#pragma unroll
    for (int i = 0; i < 2; ++i) {
      af[i].h8[0] = *(const v8h*)(&As[wm * 2 + i][0][lane][0]);
      af[i].h8[1] = *(const v8h*)(&As[wm * 2 + i][1][lane][0]);
    }
#pragma unroll
    for (int j = 0; j < 4; ++j) {
      AFrag bf;
      bf.h8[0] = *(const v8h*)(&Bs[wn * 4 + j][0][lane][0]);
      bf.h8[1] = *(const v8h*)(&Bs[wn * 4 + j][1][lane][0]);
#pragma unroll
      for (int i = 0; i < 2; ++i)
        acc[i][j].v = __builtin_amdgcn_wmma_f32_16x16x32_f16(
            false, af[i].v, false, bf.v, (short)0, acc[i][j].v, false, false);
    }
    __syncthreads();
  }

  // ---- epilogue (C/D layout: lane&15 = N, vgpr r = M (+8 for hi lanes))
#pragma unroll
  for (int i = 0; i < 2; ++i) {
#pragma unroll
    for (int j = 0; j < 4; ++j) {
#pragma unroll
      for (int r = 0; r < 8; ++r) {
        const int m    = r + ((lane >> 4) << 3);
        const int grow = m0 + wm * 32 + i * 16 + m;
        const int gcol = n0 + wn * 64 + j * 16 + (lane & 15);
        if (gcol < N) {
          const float v    = acc[i][j].f[r] + bias[gcol];
          const size_t off = (size_t)grow * N + gcol;
          if (mode == 0) {
            outF[off] = v;
          } else if (mode == 1) {
            const float g = 0.5f * v * (1.0f + erff(v * 0.70710678118654752f));
            outH[off] = (half_t)g;
          } else {
            outF[off] = resid[off] + alpha[gcol] * v;
          }
        }
      }
    }
  }
}

// ---------------------------------------------------------------------------
// center_norm: y = w * (C/(C-1) * (x - mean(x))) + b, output f16 (WMMA feed)
// one block (384 threads) per row
// ---------------------------------------------------------------------------
__global__ __launch_bounds__(384)
void center_norm_f16(const float* __restrict__ x, const float* __restrict__ w,
                     const float* __restrict__ b, half_t* __restrict__ out)
{
  __shared__ float red[512];
  const int row = blockIdx.x;
  const int tid = threadIdx.x;            // 0..383
  const float v = x[(size_t)row * DIMC + tid];
  red[tid] = v;
  if (tid < 128) red[384 + tid] = 0.0f;
  __syncthreads();
  for (int s = 256; s > 0; s >>= 1) {
    if (tid < s) red[tid] += red[tid + s];
    __syncthreads();
  }
  const float u = red[0] * (1.0f / DIMC);
  const float y = w[tid] * (((float)DIMC / (DIMC - 1.0f)) * (v - u)) + b[tid];
  out[(size_t)row * DIMC + tid] = (half_t)y;
}

// ---------------------------------------------------------------------------
// compress + window reorder: wgen[b*G+g][ (h*W_sp+w)*2+d ] = an @ cw + cb
// (K padded region of wgen is pre-zeroed by fill_h0)
// ---------------------------------------------------------------------------
__global__ __launch_bounds__(256)
void compress_kernel(const half_t* __restrict__ an, const float* __restrict__ cw,
                     const float* __restrict__ cb, half_t* __restrict__ wgen,
                     int branch)
{
  const int idx = blockIdx.x * 256 + threadIdx.x;
  if (idx >= ROWS * 16) return;
  const int token = idx >> 4;
  const int o = idx & 15;
  const int m = o >> 1, d = o & 1;
  const int b = token / NTOK;
  const int n = token - b * NTOK;
  const int hh = n / RESO, ww = n - hh * RESO;
  const half_t* base = an + (size_t)token * DIMC + branch * CBCH;
  float s = cb[o];
  for (int c = 0; c < CBCH; ++c) s += (float)base[c] * cw[c * 16 + o];
  int g, k;
  if (branch == 0) { g = (ww / 7) * HEADS + m; k = (hh * 7 + (ww % 7)) * 2 + d; }
  else             { g = (hh / 7) * HEADS + m; k = ((hh % 7) * RESO + ww) * 2 + d; }
  wgen[(size_t)(b * GWIN + g) * WDP + k] = (half_t)s;
}

// ---------------------------------------------------------------------------
// per-window softmax (over source axis) + out = v @ w, chunked over 49 cols
// one block per (b,g) window; values gathered from an_h; output f16 -> outmix
// ---------------------------------------------------------------------------
__global__ __launch_bounds__(256)
void softmax_apply_kernel(const float* __restrict__ logits,
                          const half_t* __restrict__ an,
                          half_t* __restrict__ outmix, int branch)
{
  __shared__ float  sL[WIN * JC];     // 38.4 KB
  __shared__ half_t sV[24 * WIN];     // 9.4 KB
  const int wid = blockIdx.x;
  const int b = wid >> 5;
  const int g = wid & 31;
  const int m = g & 7;
  const int tid = threadIdx.x;
  const float* Lrow = logits + (size_t)wid * NGEN;

  // gather values v[c=24][i=196]
  for (int q = tid; q < 24 * WIN; q += 256) {
    const int cc = q / WIN, i = q - cc * WIN;
    int hh, ww;
    if (branch == 0) { const int h = i / 7,    w = i - h * 7;    hh = h;                ww = (g >> 3) * 7 + w; }
    else             { const int h = i / RESO, w = i - h * RESO; hh = (g >> 3) * 7 + h; ww = w; }
    const int n = hh * RESO + ww;
    sV[cc * WIN + i] = an[((size_t)b * NTOK + n) * DIMC + branch * CBCH + m * 24 + cc];
  }
  __syncthreads();

  for (int j0 = 0; j0 < WIN; j0 += JC) {
    // load 196 x 49 logits chunk (column block)
    for (int p = tid; p < WIN * JC; p += 256) {
      const int i = p / JC, jj = p - i * JC;
      sL[i * JC + jj] = Lrow[(size_t)i * WIN + j0 + jj];
    }
    __syncthreads();
    // softmax over source axis i for each column
    if (tid < JC) {
      const int j = tid;
      float mx = -1e30f;
      for (int i = 0; i < WIN; ++i) mx = fmaxf(mx, sL[i * JC + j]);
      float s = 0.0f;
      for (int i = 0; i < WIN; ++i) {
        const float e = __expf(sL[i * JC + j] - mx);
        sL[i * JC + j] = e;
        s += e;
      }
      const float inv = 1.0f / s;
      for (int i = 0; i < WIN; ++i) sL[i * JC + j] *= inv;
    }
    __syncthreads();
    // out[c, j] = sum_i v[c,i] * w[i,j], scatter back to token layout
    for (int q = tid; q < 24 * JC; q += 256) {
      const int cc = q / JC, jj = q - cc * JC;
      float a = 0.0f;
      for (int i = 0; i < WIN; ++i)
        a += (float)sV[cc * WIN + i] * sL[i * JC + jj];
      const int j = j0 + jj;
      int hh, ww;
      if (branch == 0) { const int h = j / 7,    w = j - h * 7;    hh = h;                ww = (g >> 3) * 7 + w; }
      else             { const int h = j / RESO, w = j - h * RESO; hh = (g >> 3) * 7 + h; ww = w; }
      const int n = hh * RESO + ww;
      outmix[((size_t)b * NTOK + n) * DIMC + branch * CBCH + m * 24 + cc] = (half_t)a;
    }
    __syncthreads();
  }
}

// ---------------------------------------------------------------------------
__global__ __launch_bounds__(256)
void cvt_f2h(const float* __restrict__ in, half_t* __restrict__ out, long n)
{
  const long i = (long)blockIdx.x * 256 + threadIdx.x;
  if (i < n) out[i] = (half_t)in[i];
}

__global__ __launch_bounds__(256)
void fill_h0(half_t* __restrict__ p, long n)
{
  const long i = (long)blockIdx.x * 256 + threadIdx.x;
  if (i < n) p[i] = (half_t)0.0f;
}

// ---------------------------------------------------------------------------
extern "C" void kernel_launch(void* const* d_in, const int* in_sizes, int n_in,
                              void* d_out, int out_size, void* d_ws, size_t ws_size,
                              hipStream_t stream)
{
  (void)in_sizes; (void)n_in; (void)out_size; (void)ws_size;
  const float* x   = (const float*)d_in[0];
  const float* n1w = (const float*)d_in[1];
  const float* n1b = (const float*)d_in[2];
  const float* n2w = (const float*)d_in[3];
  const float* n2b = (const float*)d_in[4];
  const float* c0w = (const float*)d_in[5];
  const float* c0b = (const float*)d_in[6];
  const float* g0w = (const float*)d_in[7];
  const float* g0b = (const float*)d_in[8];
  const float* c1w = (const float*)d_in[9];
  const float* c1b = (const float*)d_in[10];
  const float* g1w = (const float*)d_in[11];
  const float* g1b = (const float*)d_in[12];
  const float* pw  = (const float*)d_in[13];
  const float* pb  = (const float*)d_in[14];
  const float* f1w = (const float*)d_in[15];
  const float* f1b = (const float*)d_in[16];
  const float* f2w = (const float*)d_in[17];
  const float* f2b = (const float*)d_in[18];
  const float* a1  = (const float*)d_in[19];
  const float* a2  = (const float*)d_in[20];
  float* out = (float*)d_out;

  char* ws = (char*)d_ws;
  size_t off = 0;
  auto alloc = [&](size_t bytes) -> char* {
    char* p = ws + off;
    off = (off + bytes + 255) & ~(size_t)255;
    return p;
  };
  half_t* an_h   = (half_t*)alloc((size_t)ROWS * DIMC * 2);
  half_t* wgen0  = (half_t*)alloc((size_t)GROWS * WDP * 2);
  half_t* wgen1  = (half_t*)alloc((size_t)GROWS * WDP * 2);
  half_t* gw0h   = (half_t*)alloc((size_t)WDP * NGEN * 2);
  half_t* gw1h   = (half_t*)alloc((size_t)WDP * NGEN * 2);
  half_t* pwh    = (half_t*)alloc((size_t)DIMC * DIMC * 2);
  half_t* f1wh   = (half_t*)alloc((size_t)DIMC * HID * 2);
  half_t* f2wh   = (half_t*)alloc((size_t)HID * DIMC * 2);
  half_t* outmix = (half_t*)alloc((size_t)ROWS * DIMC * 2);
  float*  xattn  = (float*)alloc((size_t)ROWS * DIMC * 4);
  half_t* mn_h   = (half_t*)alloc((size_t)ROWS * DIMC * 2);
  float*  logits = (float*)alloc((size_t)GROWS * NGEN * 4);  // 314 MB
  half_t* hbuf   = (half_t*)logits;  // MLP hidden (154 MB) reuses logits buffer

  const dim3 blk(256);

  // --- weight conversion to f16 (+ zero K padding of wgen / gw) ---
  const long nwgen = (long)GROWS * WDP;
  fill_h0<<<(unsigned)((nwgen + 255) / 256), blk, 0, stream>>>(wgen0, nwgen);
  fill_h0<<<(unsigned)((nwgen + 255) / 256), blk, 0, stream>>>(wgen1, nwgen);
  const long npad = (long)(WDP - WD) * NGEN;
  fill_h0<<<(unsigned)((npad + 255) / 256), blk, 0, stream>>>(gw0h + (size_t)WD * NGEN, npad);
  fill_h0<<<(unsigned)((npad + 255) / 256), blk, 0, stream>>>(gw1h + (size_t)WD * NGEN, npad);
  const long ngw = (long)WD * NGEN;
  cvt_f2h<<<(unsigned)((ngw + 255) / 256), blk, 0, stream>>>(g0w, gw0h, ngw);
  cvt_f2h<<<(unsigned)((ngw + 255) / 256), blk, 0, stream>>>(g1w, gw1h, ngw);
  cvt_f2h<<<(unsigned)(((long)DIMC * DIMC + 255) / 256), blk, 0, stream>>>(pw, pwh, (long)DIMC * DIMC);
  cvt_f2h<<<(unsigned)(((long)DIMC * HID + 255) / 256), blk, 0, stream>>>(f1w, f1wh, (long)DIMC * HID);
  cvt_f2h<<<(unsigned)(((long)HID * DIMC + 255) / 256), blk, 0, stream>>>(f2w, f2wh, (long)HID * DIMC);

  // --- norm1 -> an_h (f16) ---
  center_norm_f16<<<ROWS, 384, 0, stream>>>(x, n1w, n1b, an_h);

  // --- compress + window reorder for both branches ---
  compress_kernel<<<(ROWS * 16 + 255) / 256, blk, 0, stream>>>(an_h, c0w, c0b, wgen0, 0);
  compress_kernel<<<(ROWS * 16 + 255) / 256, blk, 0, stream>>>(an_h, c1w, c1b, wgen1, 1);

  // --- branch 0: generate GEMM (2048 x 416 x 38416) -> softmax+apply ---
  const dim3 ggrid((NGEN + 127) / 128, GROWS / 128);
  gemm_wmma_f16<<<ggrid, blk, 0, stream>>>(wgen0, WDP, gw0h, NGEN, g0b,
                                           logits, nullptr, nullptr, nullptr,
                                           GROWS, NGEN, WDP, 0);
  softmax_apply_kernel<<<GROWS, blk, 0, stream>>>(logits, an_h, outmix, 0);

  // --- branch 1 (reuses logits buffer; stream order guarantees safety) ---
  gemm_wmma_f16<<<ggrid, blk, 0, stream>>>(wgen1, WDP, gw1h, NGEN, g1b,
                                           logits, nullptr, nullptr, nullptr,
                                           GROWS, NGEN, WDP, 0);
  softmax_apply_kernel<<<GROWS, blk, 0, stream>>>(logits, an_h, outmix, 1);

  // --- proj + alpha1 residual: xattn = x + a1 * (outmix @ pw + pb) ---
  const dim3 pgrid(DIMC / 128, ROWS / 128);
  gemm_wmma_f16<<<pgrid, blk, 0, stream>>>(outmix, DIMC, pwh, DIMC, pb,
                                           xattn, nullptr, x, a1,
                                           ROWS, DIMC, DIMC, 2);

  // --- norm2 -> mn_h (f16) ---
  center_norm_f16<<<ROWS, 384, 0, stream>>>(xattn, n2w, n2b, mn_h);

  // --- fc1 + exact GELU -> hbuf (f16) ---
  const dim3 f1grid(HID / 128, ROWS / 128);
  gemm_wmma_f16<<<f1grid, blk, 0, stream>>>(mn_h, DIMC, f1wh, HID, f1b,
                                            nullptr, hbuf, nullptr, nullptr,
                                            ROWS, HID, DIMC, 1);

  // --- fc2 + alpha2 residual -> final output ---
  const dim3 f2grid(DIMC / 128, ROWS / 128);
  gemm_wmma_f16<<<f2grid, blk, 0, stream>>>(hbuf, HID, f2wh, DIMC, f2b,
                                            out, nullptr, xattn, a2,
                                            ROWS, DIMC, HID, 2);
}
